// ALIGNNModel_21345987461317
// MI455X (gfx1250) — matrix operand order repaired and checked by reading, not compiled, not run on hardware
//
#include <hip/hip_runtime.h>
#include <hip/hip_bf16.h>

#define NN 20000
#define EE 240000
#define LL 480000
#define BB 128
#define ATOMD 92
#define EDGED 41
#define ANGLED 1
#define HH 128
#define HQ 32

typedef __attribute__((ext_vector_type(16))) _Float16 v16h;
typedef __attribute__((ext_vector_type(8)))  float    v8f;

__device__ __forceinline__ float silu_f(float v) { return v / (1.f + __expf(-v)); }

// ---------------------------------------------------------------------------
// Tiled GEMM: C[M,N] = A[M,K] @ W[K,N] + bias, optional fused SiLU.
// 128 threads = 4 waves; each wave owns a 16-row strip of a 64-row block and
// accumulates NT N-tiles of 16 via v_wmma_f32_16x16x32_f16 (f16 in, f32 acc).
// NT is a compile-time template parameter (1,2,4,8) so the WMMA chain has no
// runtime branching. A and W converted to f16 in LDS; K zero-padded per
// 32-chunk, N zero-padded to 16.
// ---------------------------------------------------------------------------
template <int NT>
__global__ __launch_bounds__(128)
void gemm_wmma(const float* __restrict__ A, const float* __restrict__ W,
               const float* __restrict__ bias, float* __restrict__ C,
               int M, int K, int Nout, int act)
{
  constexpr int NP = NT * 16;
  __shared__ _Float16 sA[64 * 32];    // [row][k]
  __shared__ _Float16 sW[NP * 32];    // [n][k]  (W transposed)
  const int tid  = threadIdx.x;
  const int wid  = tid >> 5;
  const int lane = tid & 31;
  const int grp  = lane >> 4;
  const int l16  = lane & 15;
  const int m0   = (int)blockIdx.x * 64;
  const bool interiorM = (m0 + 64 <= M);

  v8f acc[NT] = {};

  for (int kc = 0; kc < K; kc += 32) {
    const bool fullK = (kc + 32 <= K);
    // ---- A tile: 64 rows x 32 k (f32 -> f16) ----
    if (interiorM && fullK) {
      // vectorized: 2 threads per row, 16 elems each as 4x float4
      int r = tid >> 1;
      int cb = (tid & 1) * 16;
      const float* ap = &A[(long)(m0 + r) * K + kc + cb];
      _Float16* dp = &sA[r * 32 + cb];
#pragma unroll
      for (int q = 0; q < 4; ++q) {
        float4 v = *(const float4*)(ap + q * 4);
        dp[q * 4 + 0] = (_Float16)v.x;
        dp[q * 4 + 1] = (_Float16)v.y;
        dp[q * 4 + 2] = (_Float16)v.z;
        dp[q * 4 + 3] = (_Float16)v.w;
      }
      if (kc + 32 < K)   // prefetch next chunk (global_prefetch_b8)
        __builtin_prefetch(ap + 32, 0, 1);
    } else {
      for (int idx = tid; idx < 64 * 32; idx += 128) {
        int r = idx >> 5, c = idx & 31;
        int m = m0 + r, k = kc + c;
        sA[idx] = (_Float16)((m < M && k < K) ? A[(long)m * K + k] : 0.f);
      }
    }
    // ---- W tile: 32 k x NP n, stored transposed [n][k] ----
    if (Nout == NP && fullK) {
      for (int idx = tid * 4; idx < 32 * NP; idx += 128 * 4) {
        int k = idx / NP, n = idx - k * NP;
        float4 v = *(const float4*)&W[(long)(kc + k) * Nout + n];
        sW[(n + 0) * 32 + k] = (_Float16)v.x;
        sW[(n + 1) * 32 + k] = (_Float16)v.y;
        sW[(n + 2) * 32 + k] = (_Float16)v.z;
        sW[(n + 3) * 32 + k] = (_Float16)v.w;
      }
    } else {
      for (int idx = tid; idx < NP * 32; idx += 128) {
        int n = idx >> 5, c = idx & 31;
        int k = kc + c;
        sW[idx] = (_Float16)((k < K && n < Nout) ? W[(long)k * Nout + n] : 0.f);
      }
    }
    __syncthreads();

    // A fragment (16x32 f16): lane holds row M=lane&15; group0 K 0-7/16-23,
    // group1 K 8-15/24-31 (ISA 7.12.2).
    v16h af;
    const _Float16* ap = &sA[(wid * 16 + l16) * 32];
#pragma unroll
    for (int j = 0; j < 8; ++j) {
      int kb = ((j >= 4) ? 16 : 0) + grp * 8 + (j & 3) * 2;
      af[2 * j]     = ap[kb];
      af[2 * j + 1] = ap[kb + 1];
    }
#pragma unroll
    for (int t = 0; t < NT; ++t) {
      // B fragment (32x16 f16): lane holds col N=lane&15; group0 K 0-7/16-23,
      // group1 K 8-15/24-31 (same order as A).
      v16h bf;
      const _Float16* bp = &sW[(t * 16 + l16) * 32];
#pragma unroll
      for (int j = 0; j < 8; ++j) {
        int kb = ((j >= 4) ? 16 : 0) + grp * 8 + (j & 3) * 2;
        bf[2 * j]     = bp[kb];
        bf[2 * j + 1] = bp[kb + 1];
      }
      acc[t] = __builtin_amdgcn_wmma_f32_16x16x32_f16(
          false, af, false, bf, (short)0, acc[t], false, false);
    }
    __syncthreads();
  }

  // C/D layout: lanes 0-15 -> M=r, lanes 16-31 -> M=8+r, N=lane&15.
#pragma unroll
  for (int t = 0; t < NT; ++t) {
    int n = t * 16 + l16;
    if (n < Nout) {
      float bv = bias ? bias[n] : 0.f;
#pragma unroll
      for (int r = 0; r < 8; ++r) {
        int m = m0 + wid * 16 + grp * 8 + r;
        if (m < M) {
          float v = acc[t][r] + bv;
          if (act) v = silu_f(v);
          C[(long)m * Nout + n] = v;
        }
      }
    }
  }
}

static void launch_gemm(hipStream_t s, const float* A, const float* W,
                        const float* bias, float* C, int M, int K, int Nout, int act)
{
  unsigned g = (unsigned)((M + 63) / 64);
  int NT = (Nout + 15) / 16;
  switch (NT) {
    case 1:  gemm_wmma<1><<<g, 128, 0, s>>>(A, W, bias, C, M, K, Nout, act); break;
    case 2:  gemm_wmma<2><<<g, 128, 0, s>>>(A, W, bias, C, M, K, Nout, act); break;
    case 4:  gemm_wmma<4><<<g, 128, 0, s>>>(A, W, bias, C, M, K, Nout, act); break;
    default: gemm_wmma<8><<<g, 128, 0, s>>>(A, W, bias, C, M, K, Nout, act); break;
  }
}

// ---------------------------------------------------------------------------
__global__ void zerof(float* __restrict__ p, long n) {
  long i = (long)blockIdx.x * blockDim.x + threadIdx.x;
  if (i < n) p[i] = 0.f;
}

// pre(in)=ea@eg_w+eg_b; out: pre = silu(sigmoid(sg[row]+dg[col]+pre)*ea),
// with fused per-column sum / sum-of-squares (stats[0..127], stats[128..255]).
__global__ __launch_bounds__(256)
void gate_silu_stats(const float* __restrict__ sg, const float* __restrict__ dg,
                     float* __restrict__ pre, const float* __restrict__ ea,
                     const int* __restrict__ row, const int* __restrict__ col,
                     int Ne, int ed, float* __restrict__ stats)
{
  const int tid = threadIdx.x;
  const int c   = tid & (ed - 1);     // ed is 32 or 128
  const int rg  = tid / ed;
  const int RPB = 256 / ed;
  __shared__ float red[256];
  float lsum = 0.f, lsq = 0.f;
  long base = (long)blockIdx.x * 2048;
  for (int r = rg; r < 2048; r += RPB) {
    long i = base + r;
    if (i >= Ne) break;
    long off = i * ed + c;
    float g = sg[(long)row[i] * ed + c] + dg[(long)col[i] * ed + c] + pre[off];
    g = 1.f / (1.f + __expf(-g));
    float v = silu_f(g * ea[off]);
    pre[off] = v;
    lsum += v; lsq += v * v;
  }
  red[tid] = lsum; __syncthreads();
  for (int s = 128; s >= ed; s >>= 1) { if (tid < s) red[tid] += red[tid + s]; __syncthreads(); }
  if (tid < ed) unsafeAtomicAdd(&stats[c], red[tid]);
  __syncthreads();
  red[tid] = lsq; __syncthreads();
  for (int s = 128; s >= ed; s >>= 1) { if (tid < s) red[tid] += red[tid + s]; __syncthreads(); }
  if (tid < ed) unsafeAtomicAdd(&stats[128 + c], red[tid]);
}

// buf = silu(buf) with fused column stats (node aggregation path).
__global__ __launch_bounds__(256)
void silu_stats(float* __restrict__ buf, int M, int Cc, float* __restrict__ stats)
{
  const int tid = threadIdx.x;
  const int c   = tid & (Cc - 1);
  const int rg  = tid / Cc;
  const int RPB = 256 / Cc;
  __shared__ float red[256];
  float lsum = 0.f, lsq = 0.f;
  long base = (long)blockIdx.x * 2048;
  for (int r = rg; r < 2048; r += RPB) {
    long i = base + r;
    if (i >= M) break;
    float v = silu_f(buf[i * Cc + c]);
    buf[i * Cc + c] = v;
    lsum += v; lsq += v * v;
  }
  red[tid] = lsum; __syncthreads();
  for (int s = 128; s >= Cc; s >>= 1) { if (tid < s) red[tid] += red[tid + s]; __syncthreads(); }
  if (tid < Cc) unsafeAtomicAdd(&stats[c], red[tid]);
  __syncthreads();
  red[tid] = lsq; __syncthreads();
  for (int s = 128; s >= Cc; s >>= 1) { if (tid < s) red[tid] += red[tid + s]; __syncthreads(); }
  if (tid < Cc) unsafeAtomicAdd(&stats[128 + c], red[tid]);
}

// stats: [0..127]=sum, [128..255]=sumsq -> [256..383]=scale, [384..511]=shift
__global__ void bn_finalize(float* __restrict__ stats, const float* __restrict__ g,
                            const float* __restrict__ b, int ed, float invM)
{
  int c = threadIdx.x;
  if (c < ed) {
    float mu  = stats[c] * invM;
    float var = stats[128 + c] * invM - mu * mu;
    float sc  = g[c] * rsqrtf(var + 1e-5f);
    stats[256 + c] = sc;
    stats[384 + c] = b[c] - mu * sc;
  }
}

__global__ void bn_apply_edge(float* __restrict__ pre, float* __restrict__ ea,
                              const float* __restrict__ stats, long tot, int ed)
{
  long idx = (long)blockIdx.x * blockDim.x + threadIdx.x;
  if (idx >= tot) return;
  int c = (int)(idx & (ed - 1));
  float v = pre[idx] * stats[256 + c] + stats[384 + c];
  pre[idx] = v;        // ea_new (consumed by em_w GEMM)
  ea[idx] += v;        // residual edge output
}

__global__ void bn_apply_node(float* __restrict__ x, const float* __restrict__ buf,
                              const float* __restrict__ stats, long tot, int nd)
{
  long idx = (long)blockIdx.x * blockDim.x + threadIdx.x;
  if (idx >= tot) return;
  int c = (int)(idx & (nd - 1));
  x[idx] += buf[idx] * stats[256 + c] + stats[384 + c];
}

// agg[col[i]] += sm[row[i]] + emp[i]  (segment_sum via native f32 atomics)
__global__ void scatter_add(const float* __restrict__ sm, const float* __restrict__ emp,
                            const int* __restrict__ row, const int* __restrict__ col,
                            float* __restrict__ agg, long Ne, int nd)
{
  long idx = (long)blockIdx.x * blockDim.x + threadIdx.x;
  if (idx >= Ne * nd) return;
  long i = idx / nd;
  int  c = (int)(idx - i * nd);
  unsafeAtomicAdd(&agg[(long)col[i] * nd + c], sm[(long)row[i] * nd + c] + emp[idx]);
}

__global__ void pool_add(const float* __restrict__ h, const int* __restrict__ batch,
                         float* __restrict__ pooled, float* __restrict__ cnt, int M)
{
  long idx = (long)blockIdx.x * blockDim.x + threadIdx.x;
  if (idx >= (long)M * HH) return;
  int i = (int)(idx >> 7), c = (int)(idx & 127);
  int b = batch[i];
  unsafeAtomicAdd(&pooled[b * HH + c], h[idx]);
  if (c == 0) unsafeAtomicAdd(&cnt[b], 1.f);
}

__global__ void pool_div(float* __restrict__ pooled, const float* __restrict__ cnt)
{
  int idx = blockIdx.x * blockDim.x + threadIdx.x;
  if (idx < BB * HH) pooled[idx] /= fmaxf(cnt[idx >> 7], 1.f);
}

// ---------------------------------------------------------------------------
static void run_egc(hipStream_t stream,
                    float* x, const int* row, const int* col, float* ea,
                    int n_nodes, int n_edges, int nd, int ed,
                    const float* sg_w, const float* sg_b,
                    const float* dg_w, const float* dg_b,
                    const float* eg_w, const float* eg_b,
                    const float* bne_g, const float* bne_b,
                    const float* sm_w, const float* sm_b,
                    const float* em_w, const float* em_b,
                    const float* bnn_g, const float* bnn_b,
                    float* t_sg, float* t_dg, float* t_sm,
                    float* t_pre, float* t_emp, float* t_agg, float* stats)
{
  launch_gemm(stream, x, sg_w, sg_b, t_sg, n_nodes, nd, ed, 0);
  launch_gemm(stream, x, dg_w, dg_b, t_dg, n_nodes, nd, ed, 0);
  launch_gemm(stream, ea, eg_w, eg_b, t_pre, n_edges, ed, ed, 0);
  zerof<<<1, 256, 0, stream>>>(stats, 256);
  gate_silu_stats<<<(unsigned)((n_edges + 2047) / 2048), 256, 0, stream>>>(
      t_sg, t_dg, t_pre, ea, row, col, n_edges, ed, stats);
  bn_finalize<<<1, 128, 0, stream>>>(stats, bne_g, bne_b, ed, 1.f / (float)n_edges);
  long te = (long)n_edges * ed;
  bn_apply_edge<<<(unsigned)((te + 255) / 256), 256, 0, stream>>>(t_pre, ea, stats, te, ed);
  launch_gemm(stream, x, sm_w, sm_b, t_sm, n_nodes, nd, nd, 0);
  launch_gemm(stream, t_pre, em_w, em_b, t_emp, n_edges, ed, nd, 0);
  long ta = (long)n_nodes * nd;
  zerof<<<(unsigned)((ta + 255) / 256), 256, 0, stream>>>(t_agg, ta);
  long tm = (long)n_edges * nd;
  scatter_add<<<(unsigned)((tm + 255) / 256), 256, 0, stream>>>(t_sm, t_emp, row, col,
                                                                t_agg, n_edges, nd);
  zerof<<<1, 256, 0, stream>>>(stats, 256);
  silu_stats<<<(unsigned)((n_nodes + 2047) / 2048), 256, 0, stream>>>(t_agg, n_nodes, nd, stats);
  bn_finalize<<<1, 128, 0, stream>>>(stats, bnn_g, bnn_b, nd, 1.f / (float)n_nodes);
  bn_apply_node<<<(unsigned)((ta + 255) / 256), 256, 0, stream>>>(x, t_agg, stats, ta, nd);
}

// ---------------------------------------------------------------------------
extern "C" void kernel_launch(void* const* d_in, const int* in_sizes, int n_in,
                              void* d_out, int out_size, void* d_ws, size_t ws_size,
                              hipStream_t stream)
{
  (void)in_sizes; (void)n_in; (void)out_size; (void)ws_size;
  const float* x_in  = (const float*)d_in[0];
  const int*   row   = (const int*)d_in[1];
  const int*   col   = row + EE;
  const float* eattr = (const float*)d_in[2];
  const int*   lrow  = (const int*)d_in[3];
  const int*   lcol  = lrow + LL;
  const float* lattr = (const float*)d_in[4];
  const int*   batch = (const int*)d_in[5];
  const float* W_ae = (const float*)d_in[6];  const float* b_ae = (const float*)d_in[7];
  const float* W_ee = (const float*)d_in[8];  const float* b_ee = (const float*)d_in[9];
  const float* W_ng = (const float*)d_in[10]; const float* b_ng = (const float*)d_in[11];
  const float* W1 = (const float*)d_in[12];   const float* b1 = (const float*)d_in[13];
  const float* W2 = (const float*)d_in[14];   const float* b2 = (const float*)d_in[15];
  const float* W3 = (const float*)d_in[16];   const float* b3 = (const float*)d_in[17];
  const float* lc[14]; for (int i = 0; i < 14; ++i) lc[i] = (const float*)d_in[18 + i];
  const float* ac[14]; for (int i = 0; i < 14; ++i) ac[i] = (const float*)d_in[32 + i];

  float* f = (float*)d_ws;
  size_t o = 0;
  float* h     = f + o; o += (size_t)NN * HH;   // node features
  float* e     = f + o; o += (size_t)EE * HH;   // edge features
  float* a     = f + o; o += (size_t)LL * HQ;   // angle features
  float* t_sg  = f + o; o += (size_t)EE * HQ;   // >= max(E*32, N*128)
  float* t_dg  = f + o; o += (size_t)EE * HQ;
  float* t_sm  = f + o; o += (size_t)EE * HH;   // >= max(E*128, N*128)
  float* t_pre = f + o; o += (size_t)EE * HH;   // >= max(L*32, E*128)
  float* t_emp = f + o; o += (size_t)LL * HH;   // >= max(L*128, E*128)
  float* t_agg = f + o; o += (size_t)EE * HH;   // >= max(E*128, N*128)
  float* stats = f + o; o += 512;
  float* pooled= f + o; o += (size_t)BB * HH;
  float* cnt   = f + o; o += BB;                // contiguous with pooled
  float* o1    = f + o; o += (size_t)BB * HH;
  float* o2    = f + o; o += (size_t)BB * 64;

  // embeddings
  launch_gemm(stream, x_in,  W_ae, b_ae, h, NN, ATOMD,  HH, 0);
  launch_gemm(stream, eattr, W_ee, b_ee, e, EE, EDGED,  HH, 0);
  launch_gemm(stream, lattr, W_ng, b_ng, a, LL, ANGLED, HQ, 0);

  for (int i = 0; i < 3; ++i) {
    // line-graph EGC: nodes = edge features e [E,128], edges = angles a [L,32]
    run_egc(stream, e, lrow, lcol, a, EE, LL, HH, HQ,
            lc[0] + (size_t)i * HH * HQ, lc[1] + (size_t)i * HQ,
            lc[2] + (size_t)i * HH * HQ, lc[3] + (size_t)i * HQ,
            lc[4] + (size_t)i * HQ * HQ, lc[5] + (size_t)i * HQ,
            lc[6] + (size_t)i * HQ,      lc[7] + (size_t)i * HQ,
            lc[8] + (size_t)i * HH * HH, lc[9] + (size_t)i * HH,
            lc[10] + (size_t)i * HQ * HH, lc[11] + (size_t)i * HH,
            lc[12] + (size_t)i * HH,     lc[13] + (size_t)i * HH,
            t_sg, t_dg, t_sm, t_pre, t_emp, t_agg, stats);
    // atom-graph EGC: nodes = h [N,128], edges = e [E,128]
    run_egc(stream, h, row, col, e, NN, EE, HH, HH,
            ac[0] + (size_t)i * HH * HH, ac[1] + (size_t)i * HH,
            ac[2] + (size_t)i * HH * HH, ac[3] + (size_t)i * HH,
            ac[4] + (size_t)i * HH * HH, ac[5] + (size_t)i * HH,
            ac[6] + (size_t)i * HH,      ac[7] + (size_t)i * HH,
            ac[8] + (size_t)i * HH * HH, ac[9] + (size_t)i * HH,
            ac[10] + (size_t)i * HH * HH, ac[11] + (size_t)i * HH,
            ac[12] + (size_t)i * HH,     ac[13] + (size_t)i * HH,
            t_sg, t_dg, t_sm, t_pre, t_emp, t_agg, stats);
  }
  for (int i = 3; i < 6; ++i) {
    run_egc(stream, h, row, col, e, NN, EE, HH, HH,
            ac[0] + (size_t)i * HH * HH, ac[1] + (size_t)i * HH,
            ac[2] + (size_t)i * HH * HH, ac[3] + (size_t)i * HH,
            ac[4] + (size_t)i * HH * HH, ac[5] + (size_t)i * HH,
            ac[6] + (size_t)i * HH,      ac[7] + (size_t)i * HH,
            ac[8] + (size_t)i * HH * HH, ac[9] + (size_t)i * HH,
            ac[10] + (size_t)i * HH * HH, ac[11] + (size_t)i * HH,
            ac[12] + (size_t)i * HH,     ac[13] + (size_t)i * HH,
            t_sg, t_dg, t_sm, t_pre, t_emp, t_agg, stats);
  }

  // pooling + MLP
  zerof<<<(unsigned)((BB * HH + BB + 255) / 256), 256, 0, stream>>>(pooled, BB * HH + BB);
  pool_add<<<(unsigned)(((long)NN * HH + 255) / 256), 256, 0, stream>>>(h, batch, pooled, cnt, NN);
  pool_div<<<(unsigned)((BB * HH + 255) / 256), 256, 0, stream>>>(pooled, cnt);
  launch_gemm(stream, pooled, W1, b1, o1, BB, HH, 128, 1);
  launch_gemm(stream, o1, W2, b2, o2, BB, 128, 64, 1);
  launch_gemm(stream, o2, W3, b3, (float*)d_out, BB, 64, 1, 0);
}
